// CompletePipeline_5772436046193
// MI455X (gfx1250) — compile-verified, matrix operand order checked by reading, not run on hardware
//
#include <hip/hip_runtime.h>

typedef __attribute__((ext_vector_type(16))) _Float16 v16h;
typedef __attribute__((ext_vector_type(8)))  _Float16 v8h;
typedef __attribute__((ext_vector_type(8)))  float    v8f;

#define HR 512
#define LR 256
#define NB 4

__device__ __forceinline__ int clampi(int v, int lo, int hi){ return v<lo?lo:(v>hi?hi:v); }

// k-index inside a 32-wide K chunk for the 16-bit WMMA A/B fragment layout
// (ISA 7.12.2). Per lane the 16 elements form two contiguous K-runs of 8:
// lanes 0-15: {0..7, 16..23}; lanes 16-31: {8..15, 24..31}.
__device__ __forceinline__ int frag_k(int lane, int e){
  return ((e>>3)?16:0) + ((lane&16)?8:0) + (((e>>1)&3)<<1) + (e&1);
}

// ---------------------------------------------------------------------------
// 1) bicubic 2x downsample (edge pad) + per-channel gains : x_hr -> ds (f32)
// ---------------------------------------------------------------------------
__global__ void k_down_gain(const float* __restrict__ xhr,
                            const float* __restrict__ gains,
                            float* __restrict__ ds){
  int idx = blockIdx.x*blockDim.x + threadIdx.x;
  const int total = NB*3*LR*LR;
  if(idx >= total) return;
  int ox = idx & (LR-1);
  int oy = (idx>>8) & (LR-1);
  int c  = (idx>>16) % 3;
  int b  = idx / (3*LR*LR);
  const float cw[4] = {-0.09375f, 0.59375f, 0.59375f, -0.09375f};
  const float* src = xhr + (size_t)(b*3+c)*HR*HR;
  float acc = 0.f;
  #pragma unroll
  for(int ky=0; ky<4; ky++){
    int yi = clampi(2*oy + ky - 1, 0, HR-1);
    float row = 0.f;
    #pragma unroll
    for(int kx=0; kx<4; kx++){
      int xi = clampi(2*ox + kx - 1, 0, HR-1);
      row += cw[kx] * src[yi*HR + xi];
    }
    acc += cw[ky] * row;
  }
  ds[idx] = acc * gains[c];
}

// ---------------------------------------------------------------------------
// 2) 5x5 bilateral (reflect pad) + clip + gamma : ds -> x_lr (f32, in d_out)
// ---------------------------------------------------------------------------
__global__ void k_bilateral_gamma(const float* __restrict__ ds,
                                  const float* __restrict__ sigp,
                                  const float* __restrict__ gamp,
                                  float* __restrict__ xlr){
  int idx = blockIdx.x*blockDim.x + threadIdx.x;
  const int total = NB*3*LR*LR;
  if(idx >= total) return;
  int ox = idx & (LR-1);
  int oy = (idx>>8) & (LR-1);
  int c  = (idx>>16) % 3;
  int b  = idx / (3*LR*LR);
  float sigma = sigp[0];
  float inv2s2 = 1.f/(2.f*sigma*sigma);
  float invg = 1.f/gamp[0];
  const float* img = ds + (size_t)(b*3+c)*LR*LR;
  float center = img[oy*LR + ox];
  float num = 0.f, den = 0.f;
  #pragma unroll
  for(int dy=-2; dy<=2; dy++){
    int ry = oy + dy;
    if(ry < 0) ry = -ry;
    if(ry > LR-1) ry = 2*(LR-1) - ry;
    #pragma unroll
    for(int dx=-2; dx<=2; dx++){
      int rx = ox + dx;
      if(rx < 0) rx = -rx;
      if(rx > LR-1) rx = 2*(LR-1) - rx;
      float p  = img[ry*LR + rx];
      float sw = __expf(-(float)(dy*dy + dx*dx) * 0.5f);
      float d  = p - center;
      float w  = sw * __expf(-d*d * inv2s2);
      num += w * p;
      den += w;
    }
  }
  float v = num / (den + 1e-8f);
  v = fminf(fmaxf(v, 1e-8f), 1.f);
  xlr[idx] = __powf(v, invg);
}

// ---------------------------------------------------------------------------
// 3) conv1: 3->64, 5x5, pad 2, ReLU : x_lr -> c1 (f16)  (small: VALU)
// ---------------------------------------------------------------------------
__global__ __launch_bounds__(256)
void k_conv1(const float* __restrict__ xlr, const float* __restrict__ w1,
             const float* __restrict__ b1, _Float16* __restrict__ c1out){
  int x  = threadIdx.x;          // 0..255
  int y  = blockIdx.y;           // 0..255
  int zc = blockIdx.z;           // b*64 + co
  int co = zc & 63;
  int b  = zc >> 6;
  float acc = b1[co];
  for(int ci=0; ci<3; ci++){
    const float* img = xlr + (size_t)(b*3+ci)*LR*LR;
    const float* wp  = w1 + (co*3+ci)*25;
    #pragma unroll
    for(int ky=0; ky<5; ky++){
      int yy = y + ky - 2;
      if((unsigned)yy >= LR) continue;
      #pragma unroll
      for(int kx=0; kx<5; kx++){
        int xx = x + kx - 2;
        if((unsigned)xx < LR) acc += wp[ky*5+kx] * img[yy*LR + xx];
      }
    }
  }
  acc = fmaxf(acc, 0.f);
  c1out[((size_t)zc*LR + y)*LR + x] = (_Float16)acc;
}

// ---------------------------------------------------------------------------
// 4) pack w2 (64,64,3,3) f32 -> f16 A-fragments.
//    K ordering: k = (ky*3+kx)*64 + ci  -> chunk t: r=t>>1, ci0=(t&1)*32
//    packed[((coTile*18 + t)*32 + lane)*16 + e]
// ---------------------------------------------------------------------------
__global__ void k_pack_w2(const float* __restrict__ w2, _Float16* __restrict__ pw){
  int tid = blockIdx.x*blockDim.x + threadIdx.x;
  const int total = 4*18*32*16;
  if(tid >= total) return;
  int e    = tid & 15;
  int lane = (tid>>4) & 31;
  int rest = tid >> 9;
  int t      = rest % 18;
  int coTile = rest / 18;
  int m  = lane & 15;
  int co = coTile*16 + m;
  int kl = frag_k(lane, e);            // 0..31 within chunk
  int r  = t >> 1;
  int ci = ((t & 1) << 5) + kl;
  int ky = r / 3, kx = r % 3;
  pw[tid] = (_Float16)w2[((co*64 + ci)*3 + ky)*3 + kx];
}

// ---------------------------------------------------------------------------
// 5) conv2: 64->64, 3x3, pad 1, ReLU via WMMA implicit GEMM.
//    Block = 128 thr (4 waves). Tile = 64 co x 64 px (one image row segment).
//    Input tile staged ONCE in LDS as [3 rows][66 x][72 ci-padded]; the full
//    K=576 loop is unrolled: 72 v_wmma per wave, B-fragments via 2x
//    ds_load_b128 per lane.
// ---------------------------------------------------------------------------
#define CIP 72          // padded ci stride (bank spread, keeps 16B alignment)
#define LIDX(row,xl,ci) ((((row)*66 + (xl))*CIP) + (ci))

__global__ __launch_bounds__(128)
void k_conv2_wmma(const _Float16* __restrict__ c1, const _Float16* __restrict__ pw,
                  const float* __restrict__ b2, _Float16* __restrict__ c2){
  __shared__ _Float16 ldsIn[3*66*CIP];
  int x0   = blockIdx.x * 64;
  int y    = blockIdx.y;
  int b    = blockIdx.z;
  int tid  = threadIdx.x;
  int wave = tid >> 5;
  int lane = tid & 31;

  // ---- stage input tile once: rows y-1..y+1, x0-1..x0+64, all 64 ci ----
  for(int j=tid; j<192; j+=128){
    int row = j >> 6;          // 0..2
    int ci  = j & 63;
    int ys  = y + row - 1;
    if((unsigned)ys < LR){
      const _Float16* src = c1 + (((size_t)(b*64 + ci))*LR + ys)*LR;
      // halos (zero-pad conv border)
      ldsIn[LIDX(row, 0,  ci)] = (x0 > 0)       ? src[x0-1]  : (_Float16)0.f;
      ldsIn[LIDX(row, 65, ci)] = (x0+64 < LR)   ? src[x0+64] : (_Float16)0.f;
      // interior: 8 x b128 global loads (x0*2 bytes is 128B aligned)
      const v8h* vsrc = (const v8h*)(src + x0);
      #pragma unroll
      for(int g=0; g<8; g++){
        v8h v = vsrc[g];
        #pragma unroll
        for(int e=0; e<8; e++)
          ldsIn[LIDX(row, 1 + g*8 + e, ci)] = v[e];
      }
    } else {
      for(int xl=0; xl<66; xl++)
        ldsIn[LIDX(row, xl, ci)] = (_Float16)0.f;
    }
  }
  __syncthreads();

  v8f acc[4];
  v8f zero = {};
  #pragma unroll
  for(int i=0;i<4;i++) acc[i] = zero;

  int kb = (lane & 16) ? 8 : 0;   // per-lane K-run base within chunk
  int nl = lane & 15;

  #pragma unroll
  for(int t=0; t<18; t++){
    const int r   = t >> 1;
    const int ci0 = (t & 1) << 5;
    const int ky  = r / 3, kx = r % 3;

    // per-wave A fragment: one contiguous 32B load per lane (packed weights)
    v16h a = reinterpret_cast<const v16h*>(pw)[(wave*18 + t)*32 + lane];
    if(t < 17)
      __builtin_prefetch(&pw[((size_t)(wave*18 + t + 1)*32 + lane)*16], 0, 0);

    // 4 N-tiles; reuse_a hint on all but the last (next instr reuses A)
#define WMMA_STEP(NT, RA)                                                     \
    {                                                                         \
      int xl = nl + (NT)*16 + kx;                  /* 0..65 incl. halo */     \
      const _Float16* base = &ldsIn[LIDX(ky, xl, ci0 + kb)];                  \
      v8h lo = *(const v8h*)(base);                /* K-run {kb..kb+7}   */   \
      v8h hi = *(const v8h*)(base + 16);           /* K-run {kb+16..+23} */   \
      v16h bb = __builtin_shufflevector(lo, hi,                               \
                  0,1,2,3,4,5,6,7,8,9,10,11,12,13,14,15);                     \
      acc[NT] = __builtin_amdgcn_wmma_f32_16x16x32_f16(                       \
          false, a, false, bb, (short)0, acc[NT], RA, false);                 \
    }
    WMMA_STEP(0, true)
    WMMA_STEP(1, true)
    WMMA_STEP(2, true)
    WMMA_STEP(3, false)
#undef WMMA_STEP
  }

  // epilogue: bias + ReLU, scatter per C/D VGPR layout
  int hi4 = (lane & 16) ? 8 : 0;
  #pragma unroll
  for(int nt=0; nt<4; nt++){
    int n = nl + nt*16;
    #pragma unroll
    for(int v=0; v<8; v++){
      int co = wave*16 + v + hi4;
      float val = acc[nt][v] + b2[co];
      val = fmaxf(val, 0.f);
      c2[(((size_t)(b*64 + co))*LR + y)*LR + x0 + n] = (_Float16)val;
    }
  }
}

// ---------------------------------------------------------------------------
// 6) fused bilinear x2 upsample + conv3 (64->3, 3x3, pad 1) : c2 -> x_sr
// ---------------------------------------------------------------------------
__global__ __launch_bounds__(256)
void k_up_conv3(const _Float16* __restrict__ c2, const float* __restrict__ w3,
                const float* __restrict__ b3, float* __restrict__ xsr){
  __shared__ float sw3[3*64*9];
  __shared__ float sb3[3];
  for(int i=threadIdx.x; i<3*64*9; i+=256) sw3[i] = w3[i];
  if(threadIdx.x < 3) sb3[threadIdx.x] = b3[threadIdx.x];
  __syncthreads();

  int ox = blockIdx.x*256 + threadIdx.x;   // 0..511
  int oy = blockIdx.y;
  int b  = blockIdx.z;

  int  ly[3][2], lx[3][2];
  bool vy[3], vx[3];
  #pragma unroll
  for(int d=0; d<3; d++){
    int h = oy + d - 1;
    vy[d] = (unsigned)h < HR;
    int hc = h < 0 ? 0 : h;
    int l  = hc >> 1;
    int o  = (hc & 1) ? ((l+1 < LR) ? l+1 : LR-1) : ((l > 0) ? l-1 : 0);
    ly[d][0] = l; ly[d][1] = o;
    h = ox + d - 1;
    vx[d] = (unsigned)h < HR;
    hc = h < 0 ? 0 : h;
    l  = hc >> 1;
    o  = (hc & 1) ? ((l+1 < LR) ? l+1 : LR-1) : ((l > 0) ? l-1 : 0);
    lx[d][0] = l; lx[d][1] = o;
  }

  float acc0 = sb3[0], acc1 = sb3[1], acc2 = sb3[2];

  for(int ci=0; ci<64; ci++){
    const _Float16* img = c2 + ((size_t)(b*64 + ci))*LR*LR;
    float yv[3][3];
    #pragma unroll
    for(int dy=0; dy<3; dy++){
      #pragma unroll
      for(int dx=0; dx<3; dx++){
        float v = 0.f;
        if(vy[dy] && vx[dx]){
          float r0 = 0.75f*(float)img[ly[dy][0]*LR + lx[dx][0]]
                   + 0.25f*(float)img[ly[dy][0]*LR + lx[dx][1]];
          float r1 = 0.75f*(float)img[ly[dy][1]*LR + lx[dx][0]]
                   + 0.25f*(float)img[ly[dy][1]*LR + lx[dx][1]];
          v = 0.75f*r0 + 0.25f*r1;
        }
        yv[dy][dx] = v;
      }
    }
    #pragma unroll
    for(int k=0; k<9; k++){
      float v = yv[k/3][k%3];
      acc0 += sw3[(0*64 + ci)*9 + k] * v;
      acc1 += sw3[(1*64 + ci)*9 + k] * v;
      acc2 += sw3[(2*64 + ci)*9 + k] * v;
    }
  }
  size_t base = (((size_t)b*3)*HR + oy)*HR + ox;
  xsr[base]                   = acc0;
  xsr[base + (size_t)HR*HR]   = acc1;
  xsr[base + (size_t)2*HR*HR] = acc2;
}

// ---------------------------------------------------------------------------
extern "C" void kernel_launch(void* const* d_in, const int* in_sizes, int n_in,
                              void* d_out, int out_size, void* d_ws, size_t ws_size,
                              hipStream_t stream) {
  (void)in_sizes; (void)n_in; (void)out_size; (void)ws_size;
  const float* x_hr  = (const float*)d_in[0];
  const float* gains = (const float*)d_in[1];
  const float* sigma = (const float*)d_in[2];
  const float* gamma = (const float*)d_in[3];
  const float* w1    = (const float*)d_in[4];
  const float* b1    = (const float*)d_in[5];
  const float* w2    = (const float*)d_in[6];
  const float* b2    = (const float*)d_in[7];
  const float* w3    = (const float*)d_in[8];
  const float* b3    = (const float*)d_in[9];

  float* xsr = (float*)d_out;                              // 4*3*512*512
  float* xlr = (float*)d_out + (size_t)NB*3*HR*HR;         // 4*3*256*256

  char* ws = (char*)d_ws;
  float*    ds = (float*)ws;                                           // 3,145,728 B
  _Float16* c1 = (_Float16*)(ws + 3145728);                            // 33,554,432 B
  _Float16* c2 = (_Float16*)(ws + 3145728 + 33554432);                 // 33,554,432 B
  _Float16* pw = (_Float16*)(ws + 3145728 + 33554432 + 33554432);      //     73,728 B

  const int npx = NB*3*LR*LR;     // 786432

  k_pack_w2<<<144, 256, 0, stream>>>(w2, pw);
  k_down_gain<<<(npx+255)/256, 256, 0, stream>>>(x_hr, gains, ds);
  k_bilateral_gamma<<<(npx+255)/256, 256, 0, stream>>>(ds, sigma, gamma, xlr);
  k_conv1<<<dim3(1, LR, NB*64), 256, 0, stream>>>(xlr, w1, b1, c1);
  k_conv2_wmma<<<dim3(LR/64, LR, NB), 128, 0, stream>>>(c1, pw, b2, c2);
  k_up_conv3<<<dim3(HR/256, HR, NB), 256, 0, stream>>>(c2, w3, b3, xsr);
}
